// PINO_Heston_36301063586465
// MI455X (gfx1250) — compile-verified, matrix operand order checked by reading, not run on hardware
//
#include <hip/hip_runtime.h>
#include <hip/hip_bf16.h>
#include <math.h>
#include <stdint.h>

// ---------------------------------------------------------------------------
// PINO/FNO-3D for MI455X (gfx1250).  Channel-mixing GEMMs (lift, skip MLP,
// 1x1 conv, projection) run on v_wmma_f32_16x16x32_f16 (wave32 WMMA).
// Spectral conv is implemented as exact truncated DFT matmuls (the reference
// zeroes all modes >= 12, so truncated DFTs are mathematically identical to
// rfftn/irfftn + corner masking) kept in f32 VALU.  Activation tiles are
// staged global->LDS with the Tensor Data Mover (tensor_load_to_lds +
// s_wait_tensorcnt) when the toolchain exposes the builtin.
// ---------------------------------------------------------------------------

typedef __attribute__((ext_vector_type(16))) _Float16 v16h;
typedef __attribute__((ext_vector_type(8)))  float    v8f;
typedef __attribute__((ext_vector_type(4)))  unsigned u32x4;
typedef __attribute__((ext_vector_type(4)))  int      i32x4;
typedef __attribute__((ext_vector_type(8)))  int      i32x8;

#define DEVINL __device__ __forceinline__

// problem constants
#define NB 4
#define WID 32
#define HID 128
#define SPAT 262144           // 64*64*64
#define NPTS 1048576          // NB * SPAT
#define NMODE 6912            // 24*24*12
#define SPECW 3538944         // 32*32*12*12*12*2 floats per layer

#if defined(__has_builtin)
#if __has_builtin(__builtin_amdgcn_tensor_load_to_lds) && \
    __has_builtin(__builtin_amdgcn_s_wait_tensorcnt)
#define USE_TDM 1
#endif
#endif
#ifndef USE_TDM
#define USE_TDM 0
#endif

DEVINL float gelu_f(float x) {
  return 0.5f * x * (1.0f + erff(x * 0.7071067811865475f));
}

// ---- WMMA fragment helpers (CDNA5 16-bit layouts, CDNA5 ISA) ----

// A matrix 16x32 f16, row-major in LDS with row stride rs (halfs).
// lane L (0..31): rows L&15; K pattern: j<8 -> K=(L>>4)*8+j, j>=8 -> 16+(L>>4)*8+(j-8)
DEVINL v16h load_a_lds(const _Float16* base, int rs) {
  int l  = threadIdx.x & 31;
  int r  = l & 15;
  int hb = (l >> 4) * 8;
  v16h a;
#pragma unroll
  for (int j = 0; j < 8; ++j) {
    a[j]     = base[r * rs + hb + j];
    a[j + 8] = base[r * rs + 16 + hb + j];
  }
  return a;
}

// A matrix 16x32 from an f32 x-tile staged channel-major: A[m][k] = base[k*16+m]
DEVINL v16h load_a_xtile(const float* base) {
  int l  = threadIdx.x & 31;
  int r  = l & 15;
  int hb = (l >> 4) * 8;
  v16h a;
#pragma unroll
  for (int j = 0; j < 8; ++j) {
    a[j]     = (_Float16)base[(hb + j) * 16 + r];
    a[j + 8] = (_Float16)base[(16 + hb + j) * 16 + r];
  }
  return a;
}

// B matrix 32x16 f16, pre-swizzled in global memory: one tile = 32 lanes * 16
// contiguous halfs in exact fragment order -> single 32-byte load per lane.
DEVINL v16h load_b_swz(const _Float16* tile) {
  return *(const v16h*)(tile + (threadIdx.x & 31) * 16);
}

// C/D 16x16 f32: VGPR v holds M = v + 8*(lane>>4), N = lane&15
DEVINL void store_c_half(_Float16* base, int rs, int nofs, v8f c) {
  int l  = threadIdx.x & 31;
  int n  = (l & 15) + nofs;
  int mb = (l >> 4) * 8;
#pragma unroll
  for (int v = 0; v < 8; ++v) base[(mb + v) * rs + n] = (_Float16)c[v];
}

DEVINL v8f wmma16(v16h a, v16h b, v8f c) {
  return __builtin_amdgcn_wmma_f32_16x16x32_f16(false, a, false, b,
                                                (short)0, c, false, false);
}

#if USE_TDM
// ---- Tensor Data Mover: 2-D f32 tile global -> LDS (CDNA5 ISA) ----
// D# group0: count=1 | lds_addr[63:32] | global_addr[120:64] | type=2[127:126]
// D# group1: data_size=4B, tile_dim0 x tile_dim1, tensor_dim0_stride (elems)
DEVINL void tdm_load_2d_f32(unsigned lds_byte_off, const float* gsrc,
                            int tile_x, int tile_y, int row_stride_elems) {
  unsigned long long ga = (unsigned long long)(uintptr_t)gsrc;
  u32x4 g0;
  g0[0] = 1u;                                            // one valid descriptor
  g0[1] = lds_byte_off;                                  // LDS dest (bytes)
  g0[2] = (unsigned)(ga & 0xFFFFFFFFull);                // global_addr lo
  g0[3] = (unsigned)((ga >> 32) & 0x01FFFFFFull)         // global_addr hi (57b)
        | (2u << 30);                                    // type = 2 ("image")
  i32x8 g1;
  g1[0] = 0x00020000;                                    // data_size = 4 bytes
  g1[1] = (int)(((unsigned)tile_x & 0xFFFFu) << 16);     // tensor_dim0 lo16
  g1[2] = (int)(((unsigned)tile_y & 0xFFFFu) << 16);     // tensor_dim1 lo16
  g1[3] = (int)(((unsigned)tile_x & 0xFFFFu) << 16);     // tile_dim0
  g1[4] = tile_y;                                        // tile_dim1 (dim2=0)
  g1[5] = row_stride_elems;                              // dim0_stride lo32
  g1[6] = 0;
  g1[7] = 0;
  i32x4 z4 = {};
#if __clang_major__ >= 23
  i32x8 z8 = {};
  __builtin_amdgcn_tensor_load_to_lds(g0, g1, z4, z4, z8, 0);
#else
  __builtin_amdgcn_tensor_load_to_lds(g0, g1, z4, z4, 0);
#endif
}
#endif

// ---------------------------------------------------------------------------
// Kernel: build DFT twiddle tables (15360 floats total)
// layout: wt[12][64][2] | wv[24][64][2] | ws[24][64][2] |
//         iws[64][24][2] | iwv[64][24][2] | iwt[64][12][2]
// ---------------------------------------------------------------------------
__global__ __launch_bounds__(256) void fno_build_dft(float* __restrict__ tw) {
  int id = blockIdx.x * 256 + threadIdx.x;          // 60 blocks -> 15360
  const float TP = 6.283185307179586f;
  float val = 0.f;
  if (id < 1536) {                                  // forward T (real->cplx)
    int c = id & 1, t = (id >> 1) & 63, k = id >> 7;
    float a = -TP * (float)(k * t) / 64.f;
    val = c ? sinf(a) : cosf(a);
  } else if (id < 4608) {                           // forward V
    int e = id - 1536;
    int c = e & 1, t = (e >> 1) & 63, k = e >> 7;
    int kk = (k < 12) ? k : k + 40;                 // negative-mode block
    float a = -TP * (float)(kk * t) / 64.f;
    val = c ? sinf(a) : cosf(a);
  } else if (id < 7680) {                           // forward S
    int e = id - 4608;
    int c = e & 1, t = (e >> 1) & 63, k = e >> 7;
    int kk = (k < 12) ? k : k + 40;
    float a = -TP * (float)(kk * t) / 64.f;
    val = c ? sinf(a) : cosf(a);
  } else if (id < 10752) {                          // inverse S  (x 1/64)
    int e = id - 7680;
    int c = e & 1, k = (e >> 1) % 24, s = (e >> 1) / 24;
    int kk = (k < 12) ? k : k + 40;
    float a = TP * (float)(kk * s) / 64.f;
    val = (c ? sinf(a) : cosf(a)) * (1.f / 64.f);
  } else if (id < 13824) {                          // inverse V  (x 1/64)
    int e = id - 10752;
    int c = e & 1, k = (e >> 1) % 24, s = (e >> 1) / 24;
    int kk = (k < 12) ? k : k + 40;
    float a = TP * (float)(kk * s) / 64.f;
    val = (c ? sinf(a) : cosf(a)) * (1.f / 64.f);
  } else {                                          // inverse T (cplx->real)
    int e = id - 13824;
    int c = e & 1, k = (e >> 1) % 12, t = (e >> 1) / 12;
    float sc = ((k == 0) ? 1.f : 2.f) / 64.f;       // hermitian doubling
    float a = TP * (float)(k * t) / 64.f;
    val = c ? (-sinf(a) * sc) : (cosf(a) * sc);
  }
  tw[id] = val;
}

// ---------------------------------------------------------------------------
// Kernel: convert + pre-swizzle all per-point GEMM weights to f16 fragment
// order.  hw layout (halfs):
//   lift[1024] | s1[4][4096] | s2[4][16384] | s3[4][4096] | conv[4][1024] | p1[512]
// ---------------------------------------------------------------------------
__global__ __launch_bounds__(256) void fno_convert_w(
    const float* __restrict__ lift_w, const float* __restrict__ s1w,
    const float* __restrict__ s2w,    const float* __restrict__ s3w,
    const float* __restrict__ conv_w, const float* __restrict__ p1w,
    _Float16* __restrict__ hw) {
  int id = blockIdx.x * 256 + threadIdx.x;          // 406 blocks -> 103936
  if (id >= 103936) return;
  const float* src; _Float16* dst; int rel, K, N, Kreal, trans = 0;
  if (id < 1024)        { dst = hw;          rel = id;          K = 32;  N = 32;  Kreal = 9;  src = lift_w; }
  else if (id < 17408)  { rel = id - 1024;   int L = rel / 4096;  rel %= 4096;
                          dst = hw + 1024  + L * 4096;  K = 32;  N = 128; Kreal = 9;   src = s1w  + L * 9 * 128; }
  else if (id < 82944)  { rel = id - 17408;  int L = rel / 16384; rel %= 16384;
                          dst = hw + 17408 + L * 16384; K = 128; N = 128; Kreal = 128; src = s2w  + L * 16384; }
  else if (id < 99328)  { rel = id - 82944;  int L = rel / 4096;  rel %= 4096;
                          dst = hw + 82944 + L * 4096;  K = 128; N = 32;  Kreal = 128; src = s3w  + L * 4096; }
  else if (id < 103424) { rel = id - 99328;  int L = rel / 1024;  rel %= 1024;
                          dst = hw + 99328 + L * 1024;  K = 32;  N = 32;  Kreal = 32;  src = conv_w + L * 1024; trans = 1; }
  else                  { rel = id - 103424; dst = hw + 103424;   K = 32;  N = 16;  Kreal = 32;  src = p1w; }
  int j    = rel & 15;
  int lane = (rel >> 4) & 31;
  int tile = rel >> 9;
  int ktiles = K >> 5;
  int kt = tile % ktiles, nt = tile / ktiles;
  int hb = (lane >> 4) * 8;
  int kk = (j < 8) ? (hb + j) : (16 + hb + (j - 8));
  int k = kt * 32 + kk, n = nt * 16 + (lane & 15);
  float v = 0.f;
  if (k < Kreal) v = trans ? src[n * K + k] : src[k * N + n];
  dst[rel] = (_Float16)v;
}

// ---------------------------------------------------------------------------
// Kernel: lift  x = z @ lift_w + b   (WMMA), write [B][C][S][V][T] f32
// ---------------------------------------------------------------------------
__global__ __launch_bounds__(128) void fno_lift(
    const float* __restrict__ coords, const float* __restrict__ params,
    const _Float16* __restrict__ lwh, const float* __restrict__ lb,
    float* __restrict__ xout) {
  __shared__ _Float16 sz[4][512];
  int wv = threadIdx.x >> 5, ln = threadIdx.x & 31;
  int p0 = (blockIdx.x * 4 + wv) * 16;
  for (int idx = ln; idx < 512; idx += 32) {
    int m = idx >> 5, k = idx & 31;
    int p = p0 + m, b = p >> 18;
    float zv = 0.f;
    if (k < 3)      zv = coords[(long)p * 3 + k];
    else if (k < 9) zv = params[b * 6 + (k - 3)];
    sz[wv][idx] = (_Float16)zv;
  }
  __syncthreads();
  v16h az = load_a_lds(&sz[wv][0], 32);
#pragma unroll
  for (int nt = 0; nt < 2; ++nt) {
    v8f c = {};
    c = wmma16(az, load_b_swz(lwh + nt * 512), c);
    int n = nt * 16 + (ln & 15);
    float bias = lb[n];
#pragma unroll
    for (int v = 0; v < 8; ++v) {
      int m = v + 8 * (ln >> 4);
      int p = p0 + m, b = p >> 18, sp = p & (SPAT - 1);
      xout[(b * 32 + n) * SPAT + sp] = c[v] + bias;
    }
  }
}

// ---------------------------------------------------------------------------
// Kernel: forward truncated DFT along T (real -> 12 complex modes)
// ---------------------------------------------------------------------------
__global__ __launch_bounds__(256) void fno_dft_t(
    const float* __restrict__ x, float* __restrict__ ft,
    const float* __restrict__ twg) {
  __shared__ float tw[1536];
  for (int i = threadIdx.x; i < 1536; i += 256) tw[i] = twg[i];
  __syncthreads();
  int line = blockIdx.x * 256 + threadIdx.x;        // 524288 lines
  const float* src = x + (long)line * 64;
  float re[12], im[12];
#pragma unroll
  for (int k = 0; k < 12; ++k) { re[k] = 0.f; im[k] = 0.f; }
  for (int t = 0; t < 64; ++t) {
    float xv = src[t];
#pragma unroll
    for (int k = 0; k < 12; ++k) {
      re[k] += xv * tw[(k * 64 + t) * 2];
      im[k] += xv * tw[(k * 64 + t) * 2 + 1];
    }
  }
  float* dst = ft + (long)line * 24;
#pragma unroll
  for (int k = 0; k < 12; ++k) { dst[2 * k] = re[k]; dst[2 * k + 1] = im[k]; }
}

// ---------------------------------------------------------------------------
// Kernel: generic complex truncated (i)DFT along a middle axis.
// in  viewed [n_outer][n_in ][inner] complex
// out viewed [n_outer][n_out][inner] complex
// tw  [n_out][n_in][2]  (n_out*n_in*2 <= 3072)
// ---------------------------------------------------------------------------
__global__ __launch_bounds__(256) void fno_cdft(
    const float* __restrict__ in, float* __restrict__ out,
    const float* __restrict__ twg, int n_outer, int inner, int n_in, int n_out) {
  __shared__ float tw[3072];
  int tsz = n_out * n_in * 2;
  for (int i = threadIdx.x; i < tsz; i += 256) tw[i] = twg[i];
  __syncthreads();
  int id = blockIdx.x * 256 + threadIdx.x;
  int c  = id % inner;
  int ko = (id / inner) % n_out;
  int ou = id / (inner * n_out);
  float ar = 0.f, ai = 0.f;
  const float* src = in + ((long)ou * n_in) * inner * 2 + (long)c * 2;
  const float* w   = tw + ko * n_in * 2;
  for (int j = 0; j < n_in; ++j) {
    float xr = src[0], xi = src[1];
    float wr = w[0],   wi = w[1];
    ar += xr * wr - xi * wi;
    ai += xr * wi + xi * wr;
    src += (long)inner * 2; w += 2;
  }
  float* dst = out + (((long)ou * n_out + ko) * inner + c) * 2;
  dst[0] = ar; dst[1] = ai;
}

// ---------------------------------------------------------------------------
// Kernel: spectral 32x32 complex channel mix on the 4 mode corners.
// One wave per (batch, mode-site); lane = output channel.
// ft3/gt3: [B][C][24][24][12] complex
// ---------------------------------------------------------------------------
__global__ __launch_bounds__(256) void fno_specmul(
    const float* __restrict__ ft3, float* __restrict__ gt3,
    const float* __restrict__ w1, const float* __restrict__ w2,
    const float* __restrict__ w3, const float* __restrict__ w4) {
  int ln = threadIdx.x & 31;
  int wvid = blockIdx.x * 8 + (threadIdx.x >> 5);   // 27648 waves
  int b = wvid / NMODE, m = wvid % NMODE;
  int kx = m / 288, ky = (m / 12) % 24, kz = m % 12;
  const float* w = (kx < 12) ? ((ky < 12) ? w1 : w3)
                             : ((ky < 12) ? w2 : w4);
  int m1 = (kx < 12) ? kx : kx - 12;
  int m2 = (ky < 12) ? ky : ky - 12;
  const float* xb = ft3 + ((long)(b * 32) * NMODE + m) * 2;
  __builtin_prefetch(xb, 0, 1);                     // global_prefetch_b8
  float orr = 0.f, oii = 0.f;
  for (int i = 0; i < 32; ++i) {
    float xr = xb[(long)i * NMODE * 2];
    float xi = xb[(long)i * NMODE * 2 + 1];
    int  wo  = (((i * 32 + ln) * 12 + m1) * 12 + m2) * 12 + kz;
    float wr = w[2 * wo], wi = w[2 * wo + 1];
    orr += xr * wr - xi * wi;
    oii += xr * wi + xi * wr;
  }
  float* ob = gt3 + ((long)(b * 32 + ln) * NMODE + m) * 2;
  ob[0] = orr; ob[1] = oii;
}

// ---------------------------------------------------------------------------
// Kernel: inverse truncated DFT along T (12 complex modes -> 64 reals)
// ---------------------------------------------------------------------------
__global__ __launch_bounds__(256) void fno_idft_t(
    const float* __restrict__ ft, float* __restrict__ x,
    const float* __restrict__ twg) {
  __shared__ float tw[1536];
  for (int i = threadIdx.x; i < 1536; i += 256) tw[i] = twg[i];
  __syncthreads();
  int id = blockIdx.x * 256 + threadIdx.x;          // 16.7M points
  int line = id >> 6, t = id & 63;
  const float* src = ft + (long)line * 24;
  float acc = 0.f;
#pragma unroll
  for (int k = 0; k < 12; ++k)
    acc += src[2 * k] * tw[(t * 12 + k) * 2] + src[2 * k + 1] * tw[(t * 12 + k) * 2 + 1];
  x[id] = acc;
}

// ---------------------------------------------------------------------------
// Kernel: fused per-point layer update (all WMMA):
//   skip = s3( gelu(s2( gelu(s1(z)) )) ),  x2 = x @ convW^T + cb
//   xnew = gelu(x1 + x2) + skip     (x1 = spectral output, updated in place)
// x-tile (16 points x 32 channels, f32) staged via TDM when available.
// ---------------------------------------------------------------------------
__global__ __launch_bounds__(128) void fno_pointwise(
    const float* __restrict__ coords, const float* __restrict__ params,
    const float* __restrict__ xin, float* __restrict__ xnew,
    const _Float16* __restrict__ s1wh, const float* __restrict__ s1b,
    const _Float16* __restrict__ s2wh, const float* __restrict__ s2b,
    const _Float16* __restrict__ s3wh, const float* __restrict__ s3b,
    const _Float16* __restrict__ cwh,  const float* __restrict__ cb) {
  __shared__ _Float16 sz[4][512];
  __shared__ _Float16 sh1[4][2048];
  __shared__ _Float16 sh2[4][2048];
  __shared__ float    sxf[4][512];     // x-tile, channel-major [c][m]
  int wv = threadIdx.x >> 5, ln = threadIdx.x & 31;
  int p0 = (blockIdx.x * 4 + wv) * 16;

#if USE_TDM
  {
    int b0 = p0 >> 18, sp0 = p0 & (SPAT - 1);
    unsigned lds_off = (unsigned)(uintptr_t)&sxf[wv][0];   // LDS aperture lo32
    tdm_load_2d_f32(lds_off, xin + (size_t)(b0 * 32) * SPAT + sp0,
                    16, 32, SPAT);
  }
#endif

  // stage z (pad 9 -> 32) as f16; fallback x staging if no TDM
  for (int idx = ln; idx < 512; idx += 32) {
    int m = idx >> 5, k = idx & 31;
    int p = p0 + m, b = p >> 18;
    float zv = 0.f;
    if (k < 3)      zv = coords[(long)p * 3 + k];
    else if (k < 9) zv = params[b * 6 + (k - 3)];
    sz[wv][idx] = (_Float16)zv;
#if !USE_TDM
    int mx = idx & 15, kx = idx >> 4;
    int px = p0 + mx, bx = px >> 18, spx = px & (SPAT - 1);
    sxf[wv][idx] = xin[(bx * 32 + kx) * SPAT + spx];
#endif
  }
#if USE_TDM
  __builtin_amdgcn_s_wait_tensorcnt(0);
#endif
  __syncthreads();

  // h1 = gelu(z @ s1w + b1): 8 N-tiles, K=32 (one WMMA each)
  v16h az = load_a_lds(&sz[wv][0], 32);
#pragma unroll
  for (int nt = 0; nt < 8; ++nt) {
    v8f c = {};
    c = wmma16(az, load_b_swz(s1wh + nt * 512), c);
    float bias = s1b[nt * 16 + (ln & 15)];
#pragma unroll
    for (int v = 0; v < 8; ++v) c[v] = gelu_f(c[v] + bias);
    store_c_half(&sh1[wv][0], 128, nt * 16, c);
  }
  __syncthreads();

  // h2 = gelu(h1 @ s2w + b2): 8 N-tiles x 4 K-steps
#pragma unroll
  for (int nt = 0; nt < 8; ++nt) {
    v8f c = {};
#pragma unroll
    for (int kt = 0; kt < 4; ++kt) {
      v16h a = load_a_lds(&sh1[wv][kt * 32], 128);
      c = wmma16(a, load_b_swz(s2wh + (nt * 4 + kt) * 512), c);
    }
    float bias = s2b[nt * 16 + (ln & 15)];
#pragma unroll
    for (int v = 0; v < 8; ++v) c[v] = gelu_f(c[v] + bias);
    store_c_half(&sh2[wv][0], 128, nt * 16, c);
  }
  __syncthreads();

  // skip = h2 @ s3w + b3: keep as C fragments
  v8f sk[2];
#pragma unroll
  for (int nt = 0; nt < 2; ++nt) {
    v8f c = {};
#pragma unroll
    for (int kt = 0; kt < 4; ++kt) {
      v16h a = load_a_lds(&sh2[wv][kt * 32], 128);
      c = wmma16(a, load_b_swz(s3wh + (nt * 4 + kt) * 512), c);
    }
    float bias = s3b[nt * 16 + (ln & 15)];
#pragma unroll
    for (int v = 0; v < 8; ++v) c[v] += bias;
    sk[nt] = c;
  }

  // x2 = x @ convW^T
  v16h ax = load_a_xtile(&sxf[wv][0]);
  v8f x2[2];
#pragma unroll
  for (int nt = 0; nt < 2; ++nt) {
    v8f c = {};
    x2[nt] = wmma16(ax, load_b_swz(cwh + nt * 512), c);
  }

  // combine with spectral output (in place, same thread reads+writes element)
#pragma unroll
  for (int nt = 0; nt < 2; ++nt) {
    int n = nt * 16 + (ln & 15);
    float bias = cb[n];
#pragma unroll
    for (int v = 0; v < 8; ++v) {
      int m = v + 8 * (ln >> 4);
      int p = p0 + m, b = p >> 18, sp = p & (SPAT - 1);
      long off = (long)(b * 32 + n) * SPAT + sp;
      float x1 = xnew[off];
      xnew[off] = gelu_f(x1 + x2[nt][v] + bias) + sk[nt][v];
    }
  }
}

// ---------------------------------------------------------------------------
// Kernel: projection  out = gelu(x @ p1w + b1) @ p2w + b2   (WMMA + shfl)
// ---------------------------------------------------------------------------
__global__ __launch_bounds__(128) void fno_project(
    const float* __restrict__ xin, const _Float16* __restrict__ p1wh,
    const float* __restrict__ p1b, const float* __restrict__ p2w,
    const float* __restrict__ p2b, float* __restrict__ out) {
  __shared__ float sxf[4][512];      // x-tile, channel-major [c][m]
  int wv = threadIdx.x >> 5, ln = threadIdx.x & 31;
  int p0 = (blockIdx.x * 4 + wv) * 16;
#if USE_TDM
  {
    int b0 = p0 >> 18, sp0 = p0 & (SPAT - 1);
    unsigned lds_off = (unsigned)(uintptr_t)&sxf[wv][0];
    tdm_load_2d_f32(lds_off, xin + (size_t)(b0 * 32) * SPAT + sp0,
                    16, 32, SPAT);
  }
  __builtin_amdgcn_s_wait_tensorcnt(0);
#else
  for (int idx = ln; idx < 512; idx += 32) {
    int m = idx & 15, k = idx >> 4;
    int p = p0 + m, b = p >> 18, sp = p & (SPAT - 1);
    sxf[wv][idx] = xin[(b * 32 + k) * SPAT + sp];
  }
#endif
  __syncthreads();
  v16h ax = load_a_xtile(&sxf[wv][0]);
  v8f c = {};
  c = wmma16(ax, load_b_swz(p1wh), c);
  int n = ln & 15;
  float b1 = p1b[n], w2 = p2w[n], b2 = p2b[0];
  float part[8];
#pragma unroll
  for (int v = 0; v < 8; ++v) part[v] = gelu_f(c[v] + b1) * w2;
#pragma unroll
  for (int v = 0; v < 8; ++v) {
    part[v] += __shfl_xor(part[v], 1, 32);
    part[v] += __shfl_xor(part[v], 2, 32);
    part[v] += __shfl_xor(part[v], 4, 32);
    part[v] += __shfl_xor(part[v], 8, 32);
  }
  if ((ln & 15) == 0) {
    int mb = (ln >> 4) * 8;
#pragma unroll
    for (int v = 0; v < 8; ++v) out[p0 + mb + v] = part[v] + b2;
  }
}

// ---------------------------------------------------------------------------
// Host launcher
// ---------------------------------------------------------------------------
static constexpr size_t SZ_X   = 134217728;   // 4*32*262144*4 B
static constexpr size_t OFF_XA = 0;
static constexpr size_t OFF_XB = OFF_XA + SZ_X;
static constexpr size_t OFF_F1 = OFF_XB + SZ_X;            // [BCSV][12]c : 50331648
static constexpr size_t OFF_F2 = OFF_F1 + 50331648;        // [BCS][24][12]c : 18874368
static constexpr size_t OFF_F3 = OFF_F2 + 18874368;        // [BC][24][24][12]c : 7077888
static constexpr size_t OFF_G3 = OFF_F3 + 7077888;
static constexpr size_t OFF_TW = OFF_G3 + 7077888;         // 15360 f32
static constexpr size_t OFF_HW = OFF_TW + 65536;           // 103936 f16

// twiddle sub-offsets (floats)
#define TW_WT  0
#define TW_WV  1536
#define TW_WS  4608
#define TW_IWS 7680
#define TW_IWV 10752
#define TW_IWT 13824
// half-weight sub-offsets (halfs)
#define HW_LIFT 0
#define HW_S1   1024
#define HW_S2   17408
#define HW_S3   82944
#define HW_CONV 99328
#define HW_P1   103424

extern "C" void kernel_launch(void* const* d_in, const int* in_sizes, int n_in,
                              void* d_out, int out_size, void* d_ws, size_t ws_size,
                              hipStream_t stream) {
  const float* coords = (const float*)d_in[0];
  const float* params = (const float*)d_in[1];
  const float* lift_w = (const float*)d_in[2];
  const float* lift_b = (const float*)d_in[3];
  const float* sw1    = (const float*)d_in[4];
  const float* sw2    = (const float*)d_in[5];
  const float* sw3    = (const float*)d_in[6];
  const float* sw4    = (const float*)d_in[7];
  const float* conv_w = (const float*)d_in[8];
  const float* conv_b = (const float*)d_in[9];
  const float* s1w    = (const float*)d_in[10];
  const float* s1b    = (const float*)d_in[11];
  const float* s2w    = (const float*)d_in[12];
  const float* s2b    = (const float*)d_in[13];
  const float* s3w    = (const float*)d_in[14];
  const float* s3b    = (const float*)d_in[15];
  const float* p1w    = (const float*)d_in[16];
  const float* p1b    = (const float*)d_in[17];
  const float* p2w    = (const float*)d_in[18];
  const float* p2b    = (const float*)d_in[19];

  char* ws = (char*)d_ws;
  float*     xA  = (float*)(ws + OFF_XA);
  float*     xB  = (float*)(ws + OFF_XB);
  float*     ft1 = (float*)(ws + OFF_F1);
  float*     ft2 = (float*)(ws + OFF_F2);
  float*     ft3 = (float*)(ws + OFF_F3);
  float*     gt3 = (float*)(ws + OFF_G3);
  float*     tw  = (float*)(ws + OFF_TW);
  _Float16*  hw  = (_Float16*)(ws + OFF_HW);

  fno_build_dft<<<60, 256, 0, stream>>>(tw);
  fno_convert_w<<<406, 256, 0, stream>>>(lift_w, s1w, s2w, s3w, conv_w, p1w, hw);

  fno_lift<<<16384, 128, 0, stream>>>(coords, params, hw + HW_LIFT, lift_b, xA);

  float* cur = xA;
  float* nxt = xB;
  for (int i = 0; i < 4; ++i) {
    // forward truncated DFTs: T -> V -> S
    fno_dft_t<<<2048, 256, 0, stream>>>(cur, ft1, tw + TW_WT);
    fno_cdft<<<9216, 256, 0, stream>>>(ft1, ft2, tw + TW_WV, 8192, 12, 64, 24);
    fno_cdft<<<3456, 256, 0, stream>>>(ft2, ft3, tw + TW_WS, 128, 288, 64, 24);
    // spectral channel mix on the 4 corners
    fno_specmul<<<3456, 256, 0, stream>>>(ft3, gt3,
        sw1 + (size_t)i * SPECW, sw2 + (size_t)i * SPECW,
        sw3 + (size_t)i * SPECW, sw4 + (size_t)i * SPECW);
    // inverse truncated DFTs: S -> V -> T (reusing ft2/ft1 buffers)
    fno_cdft<<<9216, 256, 0, stream>>>(gt3, ft2, tw + TW_IWS, 128, 288, 24, 64);
    fno_cdft<<<24576, 256, 0, stream>>>(ft2, ft1, tw + TW_IWV, 8192, 12, 24, 64);
    fno_idft_t<<<65536, 256, 0, stream>>>(ft1, nxt, tw + TW_IWT);
    // fused WMMA point-wise update (in place on nxt)
    fno_pointwise<<<16384, 128, 0, stream>>>(coords, params, cur, nxt,
        hw + HW_S1 + i * 4096,  s1b + i * 128,
        hw + HW_S2 + i * 16384, s2b + i * 128,
        hw + HW_S3 + i * 4096,  s3b + i * 32,
        hw + HW_CONV + i * 1024, conv_b + i * 32);
    float* t = cur; cur = nxt; nxt = t;
  }

  fno_project<<<16384, 128, 0, stream>>>(cur, hw + HW_P1, p1b, p2w, p2b,
                                         (float*)d_out);
}
